// DFConv2dNoOffset_70325794505453
// MI455X (gfx1250) — matrix operand order repaired and checked by reading, not compiled, not run on hardware
//
#include <hip/hip_runtime.h>

// Deformable conv fused sampler + bf16x3-split WMMA GEMM for gfx1250 (MI455X).
// out[b,o,p] = sum_{c,k2} weight[o,c,k2] * bilinear(x[b,c], base(p,k2)+offset[b,k2,p])
// K dimension ordered naturally as c*9+k2 so A-tiles load as contiguous b128.

#define BATCH 8
#define CIN   256
#define HH    64
#define WW    64
#define K2    9
#define COUT  256
#define NPIX  (HH*WW)
#define KTOT  (CIN*K2)      // 2304

#define OTILE 128
#define NTILE 64
#define KSTEP 32
#define KPAD  40            // halfwords per row: 80B, 16B-aligned, conflict-padded
#define TPAD  12            // dwords per tap-table entry: {w0..3, idx0..3, pad}

typedef __attribute__((ext_vector_type(16))) __bf16 v16bf;
typedef __attribute__((ext_vector_type(8)))  float  v8f;

__device__ __forceinline__ unsigned short f2bf(float f) {
  unsigned u = __float_as_uint(f);
  u += 0x7fffu + ((u >> 16) & 1u);          // round-to-nearest-even
  return (unsigned short)(u >> 16);
}
__device__ __forceinline__ float bf2f(unsigned short h) {
  return __uint_as_float(((unsigned)h) << 16);
}
__device__ __forceinline__ unsigned pack2(float a, float b) {
  return (unsigned)f2bf(a) | ((unsigned)f2bf(b) << 16);
}

__global__ __launch_bounds__(256, 2)
void dfconv_wmma_kernel(const float* __restrict__ x,
                        const float* __restrict__ offs,
                        const float* __restrict__ wgt,
                        float* __restrict__ out)
{
  __shared__ unsigned       s_tab[NTILE * K2 * TPAD];   // bilinear taps, 16B records
  __shared__ unsigned short As_hi[OTILE * KPAD];
  __shared__ unsigned short As_lo[OTILE * KPAD];
  __shared__ unsigned short Bs_hi[NTILE * KPAD];
  __shared__ unsigned short Bs_lo[NTILE * KPAD];

  const int tid = threadIdx.x;
  const int o0  = blockIdx.x * OTILE;
  const int n0  = blockIdx.y * NTILE;
  const int b   = blockIdx.z;

  // ---- Phase 0: bilinear tap tables for this pixel tile (64 pixels x 9 taps) ----
  const float* offb = offs + (size_t)b * (2 * K2) * NPIX;
  for (int t = tid; t < NTILE * K2; t += 256) {
    const int n   = t / K2;
    const int k2  = t - n * K2;
    const int pix = n0 + n;
    const int h   = pix >> 6;
    const int w   = pix & 63;
    const float dy = offb[(k2 * 2 + 0) * NPIX + pix];
    const float dx = offb[(k2 * 2 + 1) * NPIX + pix];
    const float py = (float)(h - 1 + (k2 / 3)) + dy;   // stride 1, pad 1, dil 1
    const float px = (float)(w - 1 + (k2 % 3)) + dx;
    const float y0f = floorf(py), x0f = floorf(px);
    const int   y0i = (int)y0f,   x0i = (int)x0f;
    const float fy = py - y0f,    fx = px - x0f;
    const float wts[4] = { (1.f - fy) * (1.f - fx), (1.f - fy) * fx,
                           fy * (1.f - fx),         fy * fx };
    const int cy[4] = { y0i, y0i, y0i + 1, y0i + 1 };
    const int cx[4] = { x0i, x0i + 1, x0i, x0i + 1 };
    uint4 wq, iq;
    unsigned wv[4], iv[4];
#pragma unroll
    for (int i = 0; i < 4; ++i) {
      const bool ok = (cy[i] >= 0) && (cy[i] < HH) && (cx[i] >= 0) && (cx[i] < WW);
      wv[i] = __float_as_uint(ok ? wts[i] : 0.f);
      iv[i] = ok ? (unsigned)(cy[i] * WW + cx[i]) : 0u;
    }
    wq.x = wv[0]; wq.y = wv[1]; wq.z = wv[2]; wq.w = wv[3];
    iq.x = iv[0]; iq.y = iv[1]; iq.z = iv[2]; iq.w = iv[3];
    *(uint4*)&s_tab[t * TPAD + 0] = wq;
    *(uint4*)&s_tab[t * TPAD + 4] = iq;
  }

  const int wv    = tid >> 5;
  const int lane  = tid & 31;
  const int khalf = lane >> 4;
  const int lr    = lane & 15;

  v8f acc0 = {}, acc1 = {}, acc2 = {}, acc3 = {};

  const float* xb = x + (size_t)b * CIN * NPIX;

  for (int s = 0; s < KTOT / KSTEP; ++s) {
    const int K0 = s * KSTEP;
    __syncthreads();

    // ---- Stage A tile: weight rows are contiguous in K -> b128 loads ----
#pragma unroll
    for (int i = 0; i < (OTILE * KSTEP) / (256 * 4); ++i) {
      const int e = tid + i * 256;          // 1024 float4's
      const int m = e >> 3;
      const int q = e & 7;
      const float4 f = *(const float4*)(wgt + (size_t)(o0 + m) * KTOT + K0 + q * 4);
      const float hx = bf2f(f2bf(f.x)), hy = bf2f(f2bf(f.y));
      const float hz = bf2f(f2bf(f.z)), hw = bf2f(f2bf(f.w));
      uint2 hi, lo;
      hi.x = pack2(f.x, f.y);          hi.y = pack2(f.z, f.w);
      lo.x = pack2(f.x - hx, f.y - hy); lo.y = pack2(f.z - hz, f.w - hw);
      *(uint2*)&As_hi[m * KPAD + q * 4] = hi;
      *(uint2*)&As_lo[m * KPAD + q * 4] = lo;
    }

    // ---- Stage B tile: fused bilinear gather, 32 K-slices x 64 pixels ----
#pragma unroll
    for (int i = 0; i < (NTILE * KSTEP) / 256; ++i) {
      const int e  = tid + i * 256;
      const int n  = e & 63;
      const int kk = e >> 6;
      const int K  = K0 + kk;
      const int c  = K / K2;
      const int k2 = K - c * K2;
      const unsigned* tp = &s_tab[(n * K2 + k2) * TPAD];
      const uint4 wq = *(const uint4*)(tp + 0);
      const uint4 iq = *(const uint4*)(tp + 4);
      const float* xp = xb + (size_t)c * NPIX;
      float f = __uint_as_float(wq.x) * xp[iq.x];
      f = fmaf(__uint_as_float(wq.y), xp[iq.y], f);
      f = fmaf(__uint_as_float(wq.z), xp[iq.z], f);
      f = fmaf(__uint_as_float(wq.w), xp[iq.w], f);
      const unsigned short hi = f2bf(f);
      Bs_hi[n * KPAD + kk] = hi;
      Bs_lo[n * KPAD + kk] = f2bf(f - bf2f(hi));
    }
    __syncthreads();

    // ---- A fragments (16x32 bf16): V0-3 = 8 contiguous halfwords @ khalf*8,
    //      V4-7 @ 16+khalf*8 -> two aligned b128 LDS loads each ----
    const int m = (wv << 4) + lr;
    union { v16bf v; uint4 q[2]; } ah, al;
    ah.q[0] = *(const uint4*)&As_hi[m * KPAD + khalf * 8];
    ah.q[1] = *(const uint4*)&As_hi[m * KPAD + 16 + khalf * 8];
    al.q[0] = *(const uint4*)&As_lo[m * KPAD + khalf * 8];
    al.q[1] = *(const uint4*)&As_lo[m * KPAD + 16 + khalf * 8];

    // ---- 4 N-blocks x 3 WMMA (bf16x3 split) ----
#pragma unroll
    for (int nb = 0; nb < 4; ++nb) {
      const int n = (nb << 4) + lr;
      union { v16bf v; uint4 q[2]; } bh, bl;
      // B: same K order as A: V0-3 = 8 halfwords @ khalf*8, V4-7 @ 16+khalf*8
      bh.q[0] = *(const uint4*)&Bs_hi[n * KPAD + khalf * 8];
      bh.q[1] = *(const uint4*)&Bs_hi[n * KPAD + 16 + khalf * 8];
      bl.q[0] = *(const uint4*)&Bs_lo[n * KPAD + khalf * 8];
      bl.q[1] = *(const uint4*)&Bs_lo[n * KPAD + 16 + khalf * 8];
      v8f a = (nb == 0) ? acc0 : (nb == 1) ? acc1 : (nb == 2) ? acc2 : acc3;
      a = __builtin_amdgcn_wmma_f32_16x16x32_bf16(false, ah.v, false, bh.v, (short)0, a, false, false);
      a = __builtin_amdgcn_wmma_f32_16x16x32_bf16(false, ah.v, false, bl.v, (short)0, a, false, false);
      a = __builtin_amdgcn_wmma_f32_16x16x32_bf16(false, al.v, false, bh.v, (short)0, a, false, false);
      if      (nb == 0) acc0 = a;
      else if (nb == 1) acc1 = a;
      else if (nb == 2) acc2 = a;
      else              acc3 = a;
    }
  }

  // ---- Epilogue: C/D layout — VGPR r => M = r + 8*(lane>=16), N = lane&15 ----
#pragma unroll
  for (int r = 0; r < 8; ++r) {
    const int orow = o0 + (wv << 4) + (khalf << 3) + r;
    float* op = out + ((size_t)b * COUT + orow) * NPIX + n0 + lr;
    op[0]  = acc0[r];
    op[16] = acc1[r];
    op[32] = acc2[r];
    op[48] = acc3[r];
  }
}

extern "C" void kernel_launch(void* const* d_in, const int* in_sizes, int n_in,
                              void* d_out, int out_size, void* d_ws, size_t ws_size,
                              hipStream_t stream) {
  (void)in_sizes; (void)n_in; (void)out_size; (void)d_ws; (void)ws_size;
  const float* x      = (const float*)d_in[0];
  const float* offset = (const float*)d_in[1];
  const float* weight = (const float*)d_in[2];
  float* out = (float*)d_out;

  dim3 grid(COUT / OTILE, NPIX / NTILE, BATCH);  // (2, 64, 8)
  dim3 block(256);                                // 8 wave32
  dfconv_wmma_kernel<<<grid, block, 0, stream>>>(x, offset, weight, out);
}
